// MIVolo_65197603553824
// MI455X (gfx1250) — matrix operand order repaired and checked by reading, not compiled, not run on hardware
//
#include <hip/hip_runtime.h>
#include <hip/hip_bf16.h>
#include <cstdint>

typedef _Float16 half_t;
typedef __attribute__((ext_vector_type(16))) _Float16 v16h;
typedef __attribute__((ext_vector_type(8)))  float    v8f;

union F16x16 { v16h v; uint4 q[2]; };
union H8     { uint4 q; half_t h[8]; };

__device__ __forceinline__ v8f wmma16(const F16x16 a, const F16x16 b, v8f c) {
  return __builtin_amdgcn_wmma_f32_16x16x32_f16(false, a.v, false, b.v, (short)0, c,
                                                false, false);
}

// ---------------------------------------------------------------------------
// Weight prep: f32 -> f16 (with conv reorders so A-fragments are contiguous)
// ---------------------------------------------------------------------------
__global__ void prep_f16(const float* __restrict__ s, half_t* __restrict__ d, int n) {
  int i = blockIdx.x * 256 + threadIdx.x;
  if (i < n) d[i] = (half_t)s[i];
}

// src: (64,64,3,3) OIHW -> dst [tap(9)][cout(64)][cin(64)]
__global__ void prep_conv33(const float* __restrict__ s, half_t* __restrict__ d) {
  int i = blockIdx.x * 256 + threadIdx.x;   // dst index, 9*64*64 = 36864
  if (i >= 9 * 64 * 64) return;
  int cin  = i & 63;
  int cout = (i >> 6) & 63;
  int t    = i >> 12;                        // ty*3+tx
  d[i] = (half_t)s[(cout * 64 + cin) * 9 + t];
}

// src: (192,64,8,8) OIHW -> dst [tap(64)][cout(192)][cin(64)]
__global__ void prep_projw(const float* __restrict__ s, half_t* __restrict__ d) {
  int i = blockIdx.x * 256 + threadIdx.x;   // dst index, 64*192*64 = 786432
  if (i >= 64 * 192 * 64) return;
  int cin  = i & 63;
  int cout = (i >> 6) % 192;
  int t    = i / (192 * 64);                 // ty*8+tx
  d[i] = (half_t)s[(cout * 64 + cin) * 64 + t];
}

// ---------------------------------------------------------------------------
// Stem conv1: 7x7 s1 p3, 3 -> 64 ch, NCHW f32 in, NHWC f16 out, BN+ReLU fused
// ---------------------------------------------------------------------------
__global__ __launch_bounds__(256) void conv7_bnrelu(
    const float* __restrict__ in, const float* __restrict__ w,
    const float* __restrict__ bn, half_t* __restrict__ out) {
  const int cout = threadIdx.x & 63;
  const int sub  = threadIdx.x >> 6;
  const int pix  = blockIdx.x * 4 + sub;     // 0..50175
  const int n    = blockIdx.y;
  const int y = pix / 224, x = pix % 224;
  float acc = 0.f;
  for (int ci = 0; ci < 3; ++ci) {
    const float* ip = in + (size_t)(n * 3 + ci) * (224 * 224);
    const float* wp = w + (cout * 3 + ci) * 49;
#pragma unroll
    for (int ky = 0; ky < 7; ++ky) {
      int iy = y + ky - 3;
      if (iy < 0 || iy > 223) continue;
#pragma unroll
      for (int kx = 0; kx < 7; ++kx) {
        int ix = x + kx - 3;
        if (ix < 0 || ix > 223) continue;
        acc += ip[iy * 224 + ix] * wp[ky * 7 + kx];
      }
    }
  }
  float val = acc * (bn[cout] * rsqrtf(1.f + 1e-5f)) + bn[64 + cout];
  out[((size_t)(n * 224 + y) * 224 + x) * 64 + cout] =
      (half_t)(val > 0.f ? val : 0.f);
}

// ---------------------------------------------------------------------------
// 3x3 p1 64->64 conv via WMMA implicit GEMM. NHWC f16 in/out, BN+ReLU fused.
// Block: 256 thr = 8 waves = 4 cout-tiles x 2 pixel-tiles (64 cout x 32 px).
// K = 9 taps x 64 ch, consumed as 9 x (2 chunks of 32 ch).
// LDS patch padded 64->72 halves per pixel: 144 B = 36 dwords stride, so the
// 16 lanes of a B-fragment hit 16 distinct banks (gcd(36,64)=4, cycle 16).
// ---------------------------------------------------------------------------
#define PCH 72
__global__ __launch_bounds__(256) void conv3_wmma_bnrelu(
    const half_t* __restrict__ in, const half_t* __restrict__ w,
    const float* __restrict__ bn, int stage, half_t* __restrict__ out) {
  __shared__ __align__(16) half_t patch[3][34][PCH];   // 14688 B
  const int n = blockIdx.z, y = blockIdx.y, x0 = blockIdx.x * 32;
  const int tid = threadIdx.x;

  const int wave = tid >> 5, lane = tid & 31;
  const int l = lane & 15, hi = lane >> 4;
  const int mT = wave >> 1, nT = wave & 1;
  const int M = mT * 16 + l;                 // A row = cout
  const int p = nT * 16 + l;                 // pixel within 32-tile

  // prefetch this wave's weight rows (L2 -> near caches) before the barrier
  __builtin_prefetch(w + M * 64, 0, 3);
  __builtin_prefetch(w + (4 * 64 + M) * 64, 0, 3);

  // cooperative patch load (zero-padded halo), 102 (r,c) x 8 x 16B chunks
  for (int cid = tid; cid < 816; cid += 256) {
    int sub = cid & 7, rc = cid >> 3;
    int r = rc / 34, c = rc % 34;
    int gy = y + r - 1, gx = x0 + c - 1;
    uint4 v = make_uint4(0u, 0u, 0u, 0u);
    if (gy >= 0 && gy < 224 && gx >= 0 && gx < 224)
      v = *(const uint4*)(in + (((size_t)(n * 224 + gy) * 224 + gx) * 64 + sub * 8));
    *(uint4*)(&patch[r][c][sub * 8]) = v;
  }
  __syncthreads();

  v8f acc = {};
#pragma unroll
  for (int t = 0; t < 9; ++t) {
    const int dy = t / 3, dx = t % 3;
#pragma unroll
    for (int kc = 0; kc < 2; ++kc) {
      F16x16 a, b;
      const half_t* wp = w + ((t * 64 + M) * 64 + kc * 32 + hi * 8);
      a.q[0] = *(const uint4*)(wp);
      a.q[1] = *(const uint4*)(wp + 16);
      const half_t* bp = &patch[dy][p + dx][kc * 32 + hi * 8];
      b.q[0] = *(const uint4*)(bp);
      b.q[1] = *(const uint4*)(bp + 16);
      acc = wmma16(a, b, acc);
    }
  }
  const float rs = rsqrtf(1.f + 1e-5f);
  const int px = x0 + p;
  const int cbase = mT * 16 + hi * 8;
  H8 r8;
#pragma unroll
  for (int r = 0; r < 8; ++r) {
    int cout = cbase + r;
    float val = acc[r] * (bn[stage * 128 + cout] * rs) + bn[stage * 128 + 64 + cout];
    r8.h[r] = (half_t)(val > 0.f ? val : 0.f);
  }
  *(uint4*)(out + ((size_t)(n * 224 + y) * 224 + px) * 64 + cbase) = r8.q;
}

// ---------------------------------------------------------------------------
// Proj: 8x8 s8 conv 64 -> 192 (+bias) via WMMA. NHWC f16 in, emb f16 [n*784][192].
// Block: 384 thr = 12 waves (one cout tile each), 16 pixels per block.
// ---------------------------------------------------------------------------
__global__ __launch_bounds__(384) void proj_wmma(
    const half_t* __restrict__ in, const half_t* __restrict__ w,
    const float* __restrict__ bias, half_t* __restrict__ emb) {
  const int n = blockIdx.y, pt = blockIdx.x, tid = threadIdx.x;
  const int wave = tid >> 5, l = tid & 15, hi = (tid >> 4) & 1;
  const int M = wave * 16 + l;
  const int p = pt * 16 + l;                 // 0..783
  const int py = p / 28, px = p % 28;
  const half_t* inp = in + ((size_t)(n * 224 + py * 8) * 224 + px * 8) * 64;

  v8f acc = {};
  for (int t = 0; t < 64; ++t) {             // taps ty*8+tx
    const int ty = t >> 3, tx = t & 7;
#pragma unroll
    for (int kc = 0; kc < 2; ++kc) {
      F16x16 a, b;
      const half_t* wp = w + ((size_t)(t * 192 + M) * 64 + kc * 32 + hi * 8);
      a.q[0] = *(const uint4*)(wp);
      a.q[1] = *(const uint4*)(wp + 16);
      const half_t* bp = inp + (ty * 224 + tx) * 64 + kc * 32 + hi * 8;
      b.q[0] = *(const uint4*)(bp);
      b.q[1] = *(const uint4*)(bp + 16);
      acc = wmma16(a, b, acc);
    }
  }
  const int cbase = wave * 16 + hi * 8;
  H8 r8;
#pragma unroll
  for (int r = 0; r < 8; ++r) r8.h[r] = (half_t)(acc[r] + bias[cbase + r]);
  *(uint4*)(emb + ((size_t)(n * 784) + p) * 192 + cbase) = r8.q;
}

// ---------------------------------------------------------------------------
// Shared WMMA GEMM mainloop: one 16(cout) x 16(pixel) tile, K = KD channels.
// ---------------------------------------------------------------------------
template <int KD>
__device__ __forceinline__ v8f gemm_row(const half_t* __restrict__ w,
                                        const half_t* __restrict__ xr,
                                        int M, int hi) {
  v8f acc = {};
#pragma unroll
  for (int kc = 0; kc < KD / 32; ++kc) {
    F16x16 a, b;
    const half_t* wp = w + ((size_t)M * KD + kc * 32 + hi * 8);
    a.q[0] = *(const uint4*)(wp);
    a.q[1] = *(const uint4*)(wp + 16);
    const half_t* bp = xr + kc * 32 + hi * 8;
    b.q[0] = *(const uint4*)(bp);
    b.q[1] = *(const uint4*)(bp + 16);
    acc = wmma16(a, b, acc);
  }
  return acc;
}

// q/k/v projection: W(192x192) x emb + bias -> f32 [n][192][784] (channel slabs)
__global__ __launch_bounds__(384) void gemm_qkv(
    const half_t* __restrict__ w, const float* __restrict__ bias,
    const half_t* __restrict__ x, float* __restrict__ out) {
  const int n = blockIdx.y, pt = blockIdx.x, tid = threadIdx.x;
  const int wave = tid >> 5, l = tid & 15, hi = (tid >> 4) & 1;
  const int M = wave * 16 + l;
  const int p = pt * 16 + l;
  const half_t* xr = x + ((size_t)n * 784 + p) * 192;
  v8f acc = gemm_row<192>(w, xr, M, hi);
  const int cbase = wave * 16 + hi * 8;
#pragma unroll
  for (int r = 0; r < 8; ++r)
    out[((size_t)n * 192 + cbase + r) * 784 + p] = acc[r] + bias[cbase + r];
}

// o-projection: W(192x192) x attn + bias -> f16 into cat buffer [n*784][384]
__global__ __launch_bounds__(384) void gemm_o(
    const half_t* __restrict__ w, const float* __restrict__ bias,
    const half_t* __restrict__ x, half_t* __restrict__ out, int cofs) {
  const int n = blockIdx.y, pt = blockIdx.x, tid = threadIdx.x;
  const int wave = tid >> 5, l = tid & 15, hi = (tid >> 4) & 1;
  const int M = wave * 16 + l;
  const int p = pt * 16 + l;
  const half_t* xr = x + ((size_t)n * 784 + p) * 192;
  v8f acc = gemm_row<192>(w, xr, M, hi);
  const int cbase = wave * 16 + hi * 8;
  H8 r8;
#pragma unroll
  for (int r = 0; r < 8; ++r) r8.h[r] = (half_t)(acc[r] + bias[cbase + r]);
  *(uint4*)(out + ((size_t)n * 784 + p) * 384 + cofs + cbase) = r8.q;
}

// cov: W(192x384) x cat, BN+ReLU, f32 NCHW d_out
__global__ __launch_bounds__(384) void gemm_cov(
    const half_t* __restrict__ w, const half_t* __restrict__ x,
    const float* __restrict__ g, const float* __restrict__ b,
    float* __restrict__ out) {
  const int n = blockIdx.y, pt = blockIdx.x, tid = threadIdx.x;
  const int wave = tid >> 5, l = tid & 15, hi = (tid >> 4) & 1;
  const int M = wave * 16 + l;
  const int p = pt * 16 + l;
  const half_t* xr = x + ((size_t)n * 784 + p) * 384;
  v8f acc = gemm_row<384>(w, xr, M, hi);
  const float rs = rsqrtf(1.f + 1e-5f);
  const int cbase = wave * 16 + hi * 8;
#pragma unroll
  for (int r = 0; r < 8; ++r) {
    float val = acc[r] * (g[cbase + r] * rs) + b[cbase + r];
    out[((size_t)n * 192 + cbase + r) * 784 + p] = val > 0.f ? val : 0.f;
  }
}

// ---------------------------------------------------------------------------
// Per-channel 28x28 attention (einsum keeps head & dim as batch dims):
// logits[hq,g] = sum_x q[hq,x]*k[g,x] / sqrt(28); softmax over g; out = A x V.
// One block per (n, channel). Writes NHWC f16 [n*784][192].
// ---------------------------------------------------------------------------
__global__ __launch_bounds__(256) void attn_perchan(
    const float* __restrict__ q, const float* __restrict__ k,
    const float* __restrict__ v, half_t* __restrict__ out) {
  __shared__ __align__(16) float qs[784], ks[784], vs[784], lg[784];
  const int n = blockIdx.y, c = blockIdx.x;
  const size_t bofs = ((size_t)n * 192 + c) * 784;
  {
    const float4* q4 = (const float4*)(q + bofs);
    const float4* k4 = (const float4*)(k + bofs);
    const float4* v4 = (const float4*)(v + bofs);
    for (int i = threadIdx.x; i < 196; i += 256) {      // 784/4
      *(float4*)(&qs[i * 4]) = q4[i];
      *(float4*)(&ks[i * 4]) = k4[i];
      *(float4*)(&vs[i * 4]) = v4[i];
    }
  }
  __syncthreads();
  const float scl = rsqrtf(28.f);
  for (int i = threadIdx.x; i < 784; i += 256) {
    const int hq = i / 28, g = i % 28;
    float s = 0.f;
#pragma unroll
    for (int x = 0; x < 28; ++x) s += qs[hq * 28 + x] * ks[g * 28 + x];
    lg[i] = s * scl;
  }
  __syncthreads();
  if (threadIdx.x < 28) {
    const int hq = threadIdx.x;
    float mx = -3.4e38f;
    for (int g = 0; g < 28; ++g) mx = fmaxf(mx, lg[hq * 28 + g]);
    float sum = 0.f;
    for (int g = 0; g < 28; ++g) {
      float e = __expf(lg[hq * 28 + g] - mx);
      lg[hq * 28 + g] = e; sum += e;
    }
    const float inv = 1.f / sum;
    for (int g = 0; g < 28; ++g) lg[hq * 28 + g] *= inv;
  }
  __syncthreads();
  for (int i = threadIdx.x; i < 784; i += 256) {
    const int hq = i / 28, x = i % 28;
    float s = 0.f;
#pragma unroll
    for (int g = 0; g < 28; ++g) s += lg[hq * 28 + g] * vs[g * 28 + x];
    out[((size_t)n * 784 + i) * 192 + c] = (half_t)s;
  }
}

// ---------------------------------------------------------------------------
extern "C" void kernel_launch(void* const* d_in, const int* in_sizes, int n_in,
                              void* d_out, int out_size, void* d_ws, size_t ws_size,
                              hipStream_t stream) {
  const float* face     = (const float*)d_in[0];
  const float* body     = (const float*)d_in[1];
  const float* f_w1     = (const float*)d_in[2];
  const float* f_w23    = (const float*)d_in[3];
  const float* f_bn     = (const float*)d_in[4];
  const float* f_pw     = (const float*)d_in[5];
  const float* f_pb     = (const float*)d_in[6];
  const float* b_w1     = (const float*)d_in[7];
  const float* b_w23    = (const float*)d_in[8];
  const float* b_bn     = (const float*)d_in[9];
  const float* b_pw     = (const float*)d_in[10];
  const float* b_pb     = (const float*)d_in[11];
  const float* a1_qkv_w = (const float*)d_in[12];
  const float* a1_qkv_b = (const float*)d_in[13];
  const float* a1_o_w   = (const float*)d_in[14];
  const float* a1_o_b   = (const float*)d_in[15];
  const float* a2_qkv_w = (const float*)d_in[16];
  const float* a2_qkv_b = (const float*)d_in[17];
  const float* a2_o_w   = (const float*)d_in[18];
  const float* a2_o_b   = (const float*)d_in[19];
  const float* cov_w    = (const float*)d_in[20];
  const float* cov_g    = (const float*)d_in[21];
  const float* cov_b    = (const float*)d_in[22];

  const size_t SZ_ACT = (size_t)16 * 224 * 224 * 64 * 2;  // 102,760,448 B
  const size_t SZ_EMB = (size_t)16 * 784 * 192 * 2;       //   4,816,896 B
  const size_t SZ_QKV = (size_t)16 * 192 * 784 * 4;       //   9,633,792 B

  char* base = (char*)d_ws;
  half_t* actA = (half_t*)base;
  half_t* actB = (half_t*)(base + SZ_ACT);
  char* pp = base + 2 * SZ_ACT;
  auto carve = [&](size_t nbytes) {
    char* r = pp;
    pp += (nbytes + 255) & ~(size_t)255;
    return r;
  };
  half_t* embF = (half_t*)carve(SZ_EMB);
  half_t* embB = (half_t*)carve(SZ_EMB);
  half_t* wf2  = (half_t*)carve(36864 * 2);
  half_t* wf3  = (half_t*)carve(36864 * 2);
  half_t* wb2  = (half_t*)carve(36864 * 2);
  half_t* wb3  = (half_t*)carve(36864 * 2);
  half_t* wpF  = (half_t*)carve(786432 * 2);
  half_t* wpB  = (half_t*)carve(786432 * 2);
  half_t* wq1  = (half_t*)carve(110592 * 2);
  half_t* wq2  = (half_t*)carve(110592 * 2);
  half_t* wo1  = (half_t*)carve(36864 * 2);
  half_t* wo2  = (half_t*)carve(36864 * 2);
  half_t* wcv  = (half_t*)carve(73728 * 2);
  // overlay: attention-phase buffers reuse the act region (stems are done)
  float*  qb    = (float*)base;
  float*  kb    = (float*)(base + SZ_QKV);
  float*  vb    = (float*)(base + 2 * SZ_QKV);
  half_t* attno = (half_t*)(base + 3 * SZ_QKV);
  half_t* cat   = (half_t*)(base + 3 * SZ_QKV + SZ_EMB);

  // ---- weight prep ----
  prep_conv33<<<144, 256, 0, stream>>>(f_w23, wf2);
  prep_conv33<<<144, 256, 0, stream>>>(f_w23 + 36864, wf3);
  prep_conv33<<<144, 256, 0, stream>>>(b_w23, wb2);
  prep_conv33<<<144, 256, 0, stream>>>(b_w23 + 36864, wb3);
  prep_projw<<<3072, 256, 0, stream>>>(f_pw, wpF);
  prep_projw<<<3072, 256, 0, stream>>>(b_pw, wpB);
  prep_f16<<<432, 256, 0, stream>>>(a1_qkv_w, wq1, 110592);
  prep_f16<<<432, 256, 0, stream>>>(a2_qkv_w, wq2, 110592);
  prep_f16<<<144, 256, 0, stream>>>(a1_o_w, wo1, 36864);
  prep_f16<<<144, 256, 0, stream>>>(a2_o_w, wo2, 36864);
  prep_f16<<<288, 256, 0, stream>>>(cov_w, wcv, 73728);

  // ---- face stem ----
  conv7_bnrelu<<<dim3(12544, 16), 256, 0, stream>>>(face, f_w1, f_bn, actA);
  conv3_wmma_bnrelu<<<dim3(7, 224, 16), 256, 0, stream>>>(actA, wf2, f_bn, 1, actB);
  conv3_wmma_bnrelu<<<dim3(7, 224, 16), 256, 0, stream>>>(actB, wf3, f_bn, 2, actA);
  proj_wmma<<<dim3(49, 16), 384, 0, stream>>>(actA, wpF, f_pb, embF);
  // ---- body stem ----
  conv7_bnrelu<<<dim3(12544, 16), 256, 0, stream>>>(body, b_w1, b_bn, actA);
  conv3_wmma_bnrelu<<<dim3(7, 224, 16), 256, 0, stream>>>(actA, wb2, b_bn, 1, actB);
  conv3_wmma_bnrelu<<<dim3(7, 224, 16), 256, 0, stream>>>(actB, wb3, b_bn, 2, actA);
  proj_wmma<<<dim3(49, 16), 384, 0, stream>>>(actA, wpB, b_pb, embB);

  // ---- attn1: q from face, k/v from body ----
  gemm_qkv<<<dim3(49, 16), 384, 0, stream>>>(wq1,          a1_qkv_b,       embF, qb);
  gemm_qkv<<<dim3(49, 16), 384, 0, stream>>>(wq1 + 36864,  a1_qkv_b + 192, embB, kb);
  gemm_qkv<<<dim3(49, 16), 384, 0, stream>>>(wq1 + 73728,  a1_qkv_b + 384, embB, vb);
  attn_perchan<<<dim3(192, 16), 256, 0, stream>>>(qb, kb, vb, attno);
  gemm_o<<<dim3(49, 16), 384, 0, stream>>>(wo1, a1_o_b, attno, cat, 0);

  // ---- attn2: q from body, k/v from face ----
  gemm_qkv<<<dim3(49, 16), 384, 0, stream>>>(wq2,          a2_qkv_b,       embB, qb);
  gemm_qkv<<<dim3(49, 16), 384, 0, stream>>>(wq2 + 36864,  a2_qkv_b + 192, embF, kb);
  gemm_qkv<<<dim3(49, 16), 384, 0, stream>>>(wq2 + 73728,  a2_qkv_b + 384, embF, vb);
  attn_perchan<<<dim3(192, 16), 256, 0, stream>>>(qb, kb, vb, attno);
  gemm_o<<<dim3(49, 16), 384, 0, stream>>>(wo2, a2_o_b, attno, cat, 192);

  // ---- final 1x1 conv + BN + ReLU -> d_out (NCHW f32) ----
  gemm_cov<<<dim3(49, 16), 384, 0, stream>>>(wcv, cat, cov_g, cov_b, (float*)d_out);

  (void)in_sizes; (void)n_in; (void)out_size; (void)ws_size;
}
